// Block_2559800508579
// MI455X (gfx1250) — matrix operand order repaired and checked by reading, not compiled, not run on hardware
//
#include <hip/hip_runtime.h>

// ---------------------------------------------------------------------------
// CDNA5 (gfx1250) fused GNN-style pipeline:
//   fc1 -> edge MLP (gather/concat, 2 GEMMs) -> per-det segment max (DEG=32)
//   -> det MLP (3 GEMMs) -> residual + relu
// All GEMMs use v_wmma_f32_16x16x32_f16 (f16 in, f32 accumulate).
// ---------------------------------------------------------------------------

typedef _Float16 h16_t;
typedef __attribute__((ext_vector_type(16))) _Float16 v16h;
typedef __attribute__((ext_vector_type(8)))  _Float16 v8h;
typedef __attribute__((ext_vector_type(8)))  float    v8f;

#define N_DETS 50000
#define DEG    32

// workspace layout (units: halfs). 16B alignment holds for every vector slice.
static constexpr int OFF_FC1T = 0;                       // [32][128]
static constexpr int OFF_PW1T = 4096;                    // [64][96]
static constexpr int OFF_PW2T = 10240;                   // [64][64]
static constexpr int OFF_PM1T = 14336;                   // [64][64]
static constexpr int OFF_PM2T = 18432;                   // [64][64]
static constexpr int OFF_OUTT = 22528;                   // [128][64]
static constexpr int OFF_H    = 30720;                   // h f16 [N][32]
static constexpr int OFF_M    = OFF_H + N_DETS * 32;     // m f16 [N][64]

__device__ __forceinline__ v16h cat8(v8h lo, v8h hi) {
  return __builtin_shufflevector(lo, hi, 0,1,2,3,4,5,6,7,8,9,10,11,12,13,14,15);
}

__device__ __forceinline__ v8h cvt8(const float* p) {
  v8h r;
#pragma unroll
  for (int i = 0; i < 8; ++i) r[i] = (h16_t)p[i];
  return r;
}

__device__ __forceinline__ v8f wmma16(v16h a, v16h b, v8f c) {
  // v_wmma_f32_16x16x32_f16: (neg_a, A, neg_b, B, c_mod, C, reuse_a, reuse_b)
  return __builtin_amdgcn_wmma_f32_16x16x32_f16(false, a, false, b, (short)0, c,
                                                false, false);
}

// A fragment (16x32, f16) from row-major [rows][stride] storage.
// lane l: row = mbase + (l&15); K chunks {base..base+7, base+16..base+23},
// base = kbase + 8*(l>>4).
__device__ __forceinline__ v16h load_afrag(const h16_t* S, int stride, int mbase,
                                           int kbase, int lane) {
  const h16_t* p = S + (mbase + (lane & 15)) * stride + kbase + ((lane >> 4) << 3);
  return cat8(*(const v8h*)p, *(const v8h*)(p + 16));
}

// B fragment (32x16, f16) from W^T stored row-major [N][stride]:
// lane l: column N = nbase + (l&15); K chunks {base..base+7, base+16..base+23},
// base = kbase + 8*(l>>4).
__device__ __forceinline__ v16h load_bfrag(const h16_t* Wt, int stride, int nbase,
                                           int kbase, int lane) {
  const h16_t* p = Wt + (nbase + (lane & 15)) * stride + kbase + ((lane >> 4) << 3);
  return cat8(*(const v8h*)p, *(const v8h*)(p + 16));
}

// ---------------------------------------------------------------------------
// Kernel P: transpose all fp32 weights into f16 W^T in workspace.
// ---------------------------------------------------------------------------
__global__ void prep_weights(const float* __restrict__ Wfc1,
                             const float* __restrict__ Wpw1,
                             const float* __restrict__ Wpw2,
                             const float* __restrict__ Wpm1,
                             const float* __restrict__ Wpm2,
                             const float* __restrict__ Wout,
                             h16_t* __restrict__ ws) {
  int tid = blockIdx.x * blockDim.x + threadIdx.x;
  int stp = gridDim.x * blockDim.x;
  for (int i = tid; i < 128 * 32; i += stp) {               // [128][32] -> [32][128]
    int k = i >> 5, n = i & 31;
    ws[OFF_FC1T + n * 128 + k] = (h16_t)Wfc1[i];
  }
  for (int i = tid; i < 96 * 64; i += stp) {                // [96][64] -> [64][96]
    int k = i / 64, n = i % 64;
    ws[OFF_PW1T + n * 96 + k] = (h16_t)Wpw1[i];
  }
  for (int i = tid; i < 64 * 64; i += stp) {
    int k = i >> 6, n = i & 63;
    ws[OFF_PW2T + n * 64 + k] = (h16_t)Wpw2[i];
    ws[OFF_PM1T + n * 64 + k] = (h16_t)Wpm1[i];
    ws[OFF_PM2T + n * 64 + k] = (h16_t)Wpm2[i];
  }
  for (int i = tid; i < 64 * 128; i += stp) {               // [64][128] -> [128][64]
    int k = i >> 7, n = i & 127;
    ws[OFF_OUTT + n * 64 + k] = (h16_t)Wout[i];
  }
}

// ---------------------------------------------------------------------------
// Kernel A: h = relu(detF @ W_fc1 + b_fc1), stored f16 [N][32].
// 8 waves/block, one 16-row tile per wave. 3125 tiles (exact).
// ---------------------------------------------------------------------------
__global__ void __launch_bounds__(256) fc1_kernel(const float* __restrict__ detF,
                                                  const h16_t* __restrict__ wfc1t,
                                                  const float* __restrict__ bfc1,
                                                  h16_t* __restrict__ hOut) {
  __shared__ __attribute__((aligned(16))) h16_t sW[32 * 136];
  for (int i = threadIdx.x; i < 32 * 128; i += 256) {
    int r = i >> 7, c = i & 127;
    sW[r * 136 + c] = wfc1t[i];
  }
  __syncthreads();

  int wave = threadIdx.x >> 5, lane = threadIdx.x & 31;
  int t = blockIdx.x * 8 + wave;
  if (t >= N_DETS / 16) return;
  int lh = lane & 15, lg = lane >> 4;

  v8f acc[2] = {};
  int rowg = t * 16 + lh;
#pragma unroll
  for (int ks = 0; ks < 4; ++ks) {                          // K = 128
    const float* p = detF + rowg * 128 + ks * 32 + (lg << 3);
    v16h a = cat8(cvt8(p), cvt8(p + 16));
#pragma unroll
    for (int nt = 0; nt < 2; ++nt) {
      v16h b = load_bfrag(sW, 136, nt * 16, ks * 32, lane);
      acc[nt] = wmma16(a, b, acc[nt]);
    }
  }
#pragma unroll
  for (int nt = 0; nt < 2; ++nt) {
    float bias = bfc1[nt * 16 + lh];
#pragma unroll
    for (int r = 0; r < 8; ++r) {
      float v = fmaxf(acc[nt][r] + bias, 0.0f);
      int row = t * 16 + r + (lg << 3);
      hOut[row * 32 + nt * 16 + lh] = (h16_t)v;
    }
  }
}

// ---------------------------------------------------------------------------
// Kernel B: fused edge MLP + per-detection segment max.
// 4 waves/block, one detection (32 edges) per wave. 12500 blocks (exact).
// ---------------------------------------------------------------------------
__global__ void __launch_bounds__(128) edge_kernel(const float* __restrict__ pairF,
                                                   const int* __restrict__ nIdxs,
                                                   const h16_t* __restrict__ hIn,
                                                   const h16_t* __restrict__ w1t,
                                                   const h16_t* __restrict__ w2t,
                                                   const float* __restrict__ b1,
                                                   const float* __restrict__ b2,
                                                   h16_t* __restrict__ mOut) {
  __shared__ __attribute__((aligned(16))) h16_t sW1[64 * 104];
  __shared__ __attribute__((aligned(16))) h16_t sW2[64 * 72];
  __shared__ __attribute__((aligned(16))) h16_t sX[4 * 32 * 104];

  for (int i = threadIdx.x; i < 64 * 96; i += 128) {
    int r = i / 96, c = i % 96;
    sW1[r * 104 + c] = w1t[i];
  }
  for (int i = threadIdx.x; i < 64 * 64; i += 128) {
    int r = i >> 6, c = i & 63;
    sW2[r * 72 + c] = w2t[i];
  }
  __syncthreads();

  int wave = threadIdx.x >> 5, lane = threadIdx.x & 31;
  int d = blockIdx.x * 4 + wave;                            // detection id
  int lh = lane & 15, lg = lane >> 4;
  h16_t* X = sX + wave * (32 * 104);

  // Build X[lane][0..95] = [pairFeat | h[d] | h[n] (0 if n==d)] as f16.
  {
    int e = d * DEG + lane;
    const float* pf = pairF + e * 32;
    v8h* row = (v8h*)(X + lane * 104);
    row[0] = cvt8(pf);      row[1] = cvt8(pf + 8);
    row[2] = cvt8(pf + 16); row[3] = cvt8(pf + 24);
    const v8h* hc = (const v8h*)(hIn + d * 32);
    row[4] = hc[0]; row[5] = hc[1]; row[6] = hc[2]; row[7] = hc[3];
    int n = nIdxs[e];
    if (n == d) {
      v8h z = {};
      row[8] = z; row[9] = z; row[10] = z; row[11] = z;
    } else {
      const v8h* hn = (const v8h*)(hIn + n * 32);
      row[8] = hn[0]; row[9] = hn[1]; row[10] = hn[2]; row[11] = hn[3];
    }
  }

  // ---- Layer 1: [32,96] x [96,64] ----
  v16h a1[2][3];
#pragma unroll
  for (int mt = 0; mt < 2; ++mt)
#pragma unroll
    for (int ks = 0; ks < 3; ++ks)
      a1[mt][ks] = load_afrag(X, 104, mt * 16, ks * 32, lane);

  v8f acc1[2][4] = {};
#pragma unroll
  for (int ks = 0; ks < 3; ++ks)
#pragma unroll
    for (int nt = 0; nt < 4; ++nt) {
      v16h b = load_bfrag(sW1, 104, nt * 16, ks * 32, lane);
      acc1[0][nt] = wmma16(a1[0][ks], b, acc1[0][nt]);
      acc1[1][nt] = wmma16(a1[1][ks], b, acc1[1][nt]);
    }

  // bias + relu -> reuse X as Y [32][64] (A frags already in registers).
#pragma unroll
  for (int mt = 0; mt < 2; ++mt)
#pragma unroll
    for (int nt = 0; nt < 4; ++nt) {
      float bias = b1[nt * 16 + lh];
#pragma unroll
      for (int r = 0; r < 8; ++r) {
        float v = fmaxf(acc1[mt][nt][r] + bias, 0.0f);
        X[(mt * 16 + r + (lg << 3)) * 104 + nt * 16 + lh] = (h16_t)v;
      }
    }

  // ---- Layer 2: [32,64] x [64,64] ----
  v16h a2[2][2];
#pragma unroll
  for (int mt = 0; mt < 2; ++mt)
#pragma unroll
    for (int ks = 0; ks < 2; ++ks)
      a2[mt][ks] = load_afrag(X, 104, mt * 16, ks * 32, lane);

  v8f acc2[2][4] = {};
#pragma unroll
  for (int ks = 0; ks < 2; ++ks)
#pragma unroll
    for (int nt = 0; nt < 4; ++nt) {
      v16h b = load_bfrag(sW2, 72, nt * 16, ks * 32, lane);
      acc2[0][nt] = wmma16(a2[0][ks], b, acc2[0][nt]);
      acc2[1][nt] = wmma16(a2[1][ks], b, acc2[1][nt]);
    }

  // bias + relu + max over the detection's 32 edges; lanes l/l+16 hold
  // the same output column (rows r vs r+8) -> one shfl_xor(16) finishes it.
#pragma unroll
  for (int nt = 0; nt < 4; ++nt) {
    float bias = b2[nt * 16 + lh];
    float mv = 0.0f;                                        // relu output >= 0
#pragma unroll
    for (int mt = 0; mt < 2; ++mt)
#pragma unroll
      for (int r = 0; r < 8; ++r)
        mv = fmaxf(mv, fmaxf(acc2[mt][nt][r] + bias, 0.0f));
    mv = fmaxf(mv, __shfl_xor(mv, 16, 32));
    if (lane < 16) mOut[d * 64 + nt * 16 + lane] = (h16_t)mv;
  }
}

// ---------------------------------------------------------------------------
// Kernel C: m -> pm1 -> pm2 -> out, residual + relu. 8 waves/block,
// one 16-detection tile per wave. 3125 tiles (exact).
// ---------------------------------------------------------------------------
__global__ void __launch_bounds__(256) detmlp_kernel(const h16_t* __restrict__ mIn,
                                                     const float* __restrict__ detF,
                                                     const h16_t* __restrict__ pm1t,
                                                     const h16_t* __restrict__ pm2t,
                                                     const h16_t* __restrict__ outt,
                                                     const float* __restrict__ bpm1,
                                                     const float* __restrict__ bpm2,
                                                     const float* __restrict__ bout,
                                                     float* __restrict__ out) {
  __shared__ __attribute__((aligned(16))) h16_t sP1[64 * 72];
  __shared__ __attribute__((aligned(16))) h16_t sP2[64 * 72];
  __shared__ __attribute__((aligned(16))) h16_t sWo[128 * 72];
  __shared__ __attribute__((aligned(16))) h16_t sY[8 * 16 * 72];

  for (int i = threadIdx.x; i < 64 * 64; i += 256) {
    int r = i >> 6, c = i & 63;
    sP1[r * 72 + c] = pm1t[i];
    sP2[r * 72 + c] = pm2t[i];
  }
  for (int i = threadIdx.x; i < 128 * 64; i += 256) {
    int r = i >> 6, c = i & 63;
    sWo[r * 72 + c] = outt[i];
  }
  __syncthreads();

  int wave = threadIdx.x >> 5, lane = threadIdx.x & 31;
  int t = blockIdx.x * 8 + wave;
  if (t >= N_DETS / 16) return;
  int lh = lane & 15, lg = lane >> 4;
  h16_t* Y = sY + wave * (16 * 72);

  // ---- pm1: A straight from global m (f16 row-major). ----
  v16h a[2];
#pragma unroll
  for (int ks = 0; ks < 2; ++ks) {
    const h16_t* p = mIn + (t * 16 + lh) * 64 + ks * 32 + (lg << 3);
    a[ks] = cat8(*(const v8h*)p, *(const v8h*)(p + 16));
  }
  v8f acc[4] = {};
#pragma unroll
  for (int ks = 0; ks < 2; ++ks)
#pragma unroll
    for (int nt = 0; nt < 4; ++nt)
      acc[nt] = wmma16(a[ks], load_bfrag(sP1, 72, nt * 16, ks * 32, lane), acc[nt]);
#pragma unroll
  for (int nt = 0; nt < 4; ++nt) {
    float bias = bpm1[nt * 16 + lh];
#pragma unroll
    for (int r = 0; r < 8; ++r)
      Y[(r + (lg << 3)) * 72 + nt * 16 + lh] =
          (h16_t)fmaxf(acc[nt][r] + bias, 0.0f);
  }

  // ---- pm2 ----
#pragma unroll
  for (int ks = 0; ks < 2; ++ks) a[ks] = load_afrag(Y, 72, 0, ks * 32, lane);
  v8f acc2[4] = {};
#pragma unroll
  for (int ks = 0; ks < 2; ++ks)
#pragma unroll
    for (int nt = 0; nt < 4; ++nt)
      acc2[nt] = wmma16(a[ks], load_bfrag(sP2, 72, nt * 16, ks * 32, lane), acc2[nt]);
#pragma unroll
  for (int nt = 0; nt < 4; ++nt) {
    float bias = bpm2[nt * 16 + lh];
#pragma unroll
    for (int r = 0; r < 8; ++r)
      Y[(r + (lg << 3)) * 72 + nt * 16 + lh] =
          (h16_t)fmaxf(acc2[nt][r] + bias, 0.0f);
  }

  // ---- out: [16,64] x [64,128], then residual + relu. ----
#pragma unroll
  for (int ks = 0; ks < 2; ++ks) a[ks] = load_afrag(Y, 72, 0, ks * 32, lane);
  v8f acco[8] = {};
#pragma unroll
  for (int ks = 0; ks < 2; ++ks)
#pragma unroll
    for (int nt = 0; nt < 8; ++nt)
      acco[nt] = wmma16(a[ks], load_bfrag(sWo, 72, nt * 16, ks * 32, lane), acco[nt]);
#pragma unroll
  for (int nt = 0; nt < 8; ++nt) {
    float bias = bout[nt * 16 + lh];
#pragma unroll
    for (int r = 0; r < 8; ++r) {
      int row = t * 16 + r + (lg << 3);
      int col = nt * 16 + lh;
      float v = acco[nt][r] + bias + detF[row * 128 + col];
      out[row * 128 + col] = fmaxf(v, 0.0f);
    }
  }
}

// ---------------------------------------------------------------------------
extern "C" void kernel_launch(void* const* d_in, const int* in_sizes, int n_in,
                              void* d_out, int out_size, void* d_ws, size_t ws_size,
                              hipStream_t stream) {
  const float* detF  = (const float*)d_in[0];
  // d_in[1] = cIdxs: repeat(arange(N), DEG) -> derived analytically, never read.
  const int*   nIdxs = (const int*)d_in[2];
  const float* pairF = (const float*)d_in[3];
  const float* Wfc1 = (const float*)d_in[4];  const float* bfc1 = (const float*)d_in[5];
  const float* Wpw1 = (const float*)d_in[6];  const float* bpw1 = (const float*)d_in[7];
  const float* Wpw2 = (const float*)d_in[8];  const float* bpw2 = (const float*)d_in[9];
  const float* Wpm1 = (const float*)d_in[10]; const float* bpm1 = (const float*)d_in[11];
  const float* Wpm2 = (const float*)d_in[12]; const float* bpm2 = (const float*)d_in[13];
  const float* Wout = (const float*)d_in[14]; const float* bout = (const float*)d_in[15];
  float* out = (float*)d_out;
  h16_t* ws = (h16_t*)d_ws;

  prep_weights<<<64, 256, 0, stream>>>(Wfc1, Wpw1, Wpw2, Wpm1, Wpm2, Wout, ws);

  int fc1_blocks = (N_DETS / 16 + 7) / 8;                   // 391
  fc1_kernel<<<fc1_blocks, 256, 0, stream>>>(detF, ws + OFF_FC1T, bfc1, ws + OFF_H);

  edge_kernel<<<N_DETS / 4, 128, 0, stream>>>(pairF, nIdxs, ws + OFF_H,
                                              ws + OFF_PW1T, ws + OFF_PW2T,
                                              bpw1, bpw2, ws + OFF_M);

  detmlp_kernel<<<fc1_blocks, 256, 0, stream>>>(ws + OFF_M, detF,
                                                ws + OFF_PM1T, ws + OFF_PM2T,
                                                ws + OFF_OUTT, bpm1, bpm2, bout, out);
}
